// SparseMoELayer_53833120088475
// MI455X (gfx1250) — matrix operand order repaired and checked by reading, not compiled, not run on hardware
//
#include <hip/hip_runtime.h>
#include <hip/hip_bf16.h>

typedef __attribute__((ext_vector_type(16))) _Float16 v16h;
typedef __attribute__((ext_vector_type(8)))  _Float16 v8h;
typedef __attribute__((ext_vector_type(8)))  float    v8f;

#define D_IN   256
#define D_OUT  256
#define NEXP   8
#define M_TILE 128
#define K_TILE 64
#define LDA    72   // half-element row stride in LDS (144 B, 16B-aligned, conflict-skewed)
#define NKT    (D_IN / K_TILE)

union V16 { v16h v; v8h h[2]; };

#define WMMA_F16(a, b, c) \
  __builtin_amdgcn_wmma_f32_16x16x32_f16(false, (a), false, (b), (short)0, (c), false, false)

// ---------------------------------------------------------------------------
// prep: expert_w f32 -> f16 copy; bias_proj[e][o] = dot(expert_bias[e], W[e][o])
// ---------------------------------------------------------------------------
__global__ void moe_prep(const float* __restrict__ expert_w,
                         const float* __restrict__ expert_bias,
                         _Float16* __restrict__ wh,
                         float* __restrict__ bias_proj) {
  int eo = blockIdx.x;            // e*D_OUT + o
  int e  = eo >> 8;
  int i  = threadIdx.x;           // 0..255
  float w = expert_w[(size_t)eo * D_IN + i];
  wh[(size_t)eo * D_IN + i] = (_Float16)w;
  __shared__ float red[256];
  red[i] = w * expert_bias[e * D_IN + i];
  __syncthreads();
  for (int s = 128; s > 0; s >>= 1) {
    if (i < s) red[i] += red[i + s];
    __syncthreads();
  }
  if (i == 0) bias_proj[eo] = red[0];
}

// ---------------------------------------------------------------------------
// gating: logits = x @ w_gate, top-2, softmax over the two kept logits,
// atomic-append (token, gate) into per-expert lists.
// ---------------------------------------------------------------------------
__global__ void moe_gate(const float* __restrict__ x,
                         const float* __restrict__ w_gate,
                         int*   __restrict__ counts,
                         int*   __restrict__ tok_idx,
                         float* __restrict__ gate_arr,
                         int n_tok) {
  __shared__ float wg[D_IN * NEXP];     // 8 KB, broadcast reads
  int tid = threadIdx.x;
  for (int i = tid; i < D_IN * NEXP; i += 256) wg[i] = w_gate[i];
  __syncthreads();

  int n = blockIdx.x * 256 + tid;
  const float* xr = x + (size_t)n * D_IN;
  float acc[NEXP];
  #pragma unroll
  for (int e = 0; e < NEXP; ++e) acc[e] = 0.f;
  for (int i = 0; i < D_IN; ++i) {
    float xi = xr[i];
    #pragma unroll
    for (int e = 0; e < NEXP; ++e) acc[e] += xi * wg[i * NEXP + e];
  }

  int e0 = 0; float v0 = acc[0];
  #pragma unroll
  for (int e = 1; e < NEXP; ++e) if (acc[e] > v0) { v0 = acc[e]; e0 = e; }
  int e1 = -1; float v1 = -3.4e38f;
  #pragma unroll
  for (int e = 0; e < NEXP; ++e) if (e != e0 && acc[e] > v1) { v1 = acc[e]; e1 = e; }

  float g0 = 1.f / (1.f + __expf(v1 - v0));   // softmax over {v0, v1}, v0 = max
  float g1 = 1.f - g0;

  int p0 = atomicAdd(&counts[e0], 1);
  tok_idx[(size_t)e0 * n_tok + p0]  = n;
  gate_arr[(size_t)e0 * n_tok + p0] = g0;
  int p1 = atomicAdd(&counts[e1], 1);
  tok_idx[(size_t)e1 * n_tok + p1]  = n;
  gate_arr[(size_t)e1 * n_tok + p1] = g1;
}

// ---------------------------------------------------------------------------
// expert GEMM, 512 threads = 16 waves. Tile: 128 gathered tokens x 256 cols.
//   wave -> (mg = wave>>2 : 2 M sub-tiles, ng = wave&3 : 4 N sub-tiles)
//   per wave per kk-step: 6 named fragments (a0,a1,b0..b3) loaded up-front,
//   then 8 v_wmma -> 12 ds_load_b128 per 8 WMMA, pipelined dscnt waits.
// Staging:
//   A (gathered x, f32->f16 cvt) : global->VGPR prefetch, ds_store, as before
//   B (f16 weights, pure copy)   : GLOBAL_LOAD_ASYNC_TO_LDS_B128, ASYNCcnt,
//                                  double-buffered LDS so kt+1's transfer
//                                  overlaps kt's WMMAs without VGPR traffic
// ---------------------------------------------------------------------------
__global__ __launch_bounds__(512) void moe_gemm(
    const float*    __restrict__ x,
    const _Float16* __restrict__ wh,
    const float*    __restrict__ bias_proj,
    const int*      __restrict__ counts,
    const int*      __restrict__ tok_idx,
    const float*    __restrict__ gate_arr,
    float*          __restrict__ out,
    int n_tok) {
  int e = blockIdx.y;
  int cnt = counts[e];
  int row_base = blockIdx.x * M_TILE;
  if (row_base >= cnt) return;

  __shared__ __align__(16) _Float16 As[M_TILE][LDA];      // 18 KB
  __shared__ __align__(16) _Float16 Bs[2][D_OUT][LDA];    // 2 x 36.9 KB
  __shared__ int   tok_s[M_TILE];
  __shared__ float gate_s[M_TILE];

  int tid = threadIdx.x;
  if (tid < M_TILE) {
    int r = row_base + tid;
    if (r < cnt) {
      tok_s[tid]  = tok_idx[(size_t)e * n_tok + r];
      gate_s[tid] = gate_arr[(size_t)e * n_tok + r];
    } else { tok_s[tid] = 0; gate_s[tid] = 0.f; }
  }
  __syncthreads();

  // staging assignment (512 threads)
  int arow = tid >> 2;                 // 0..127, x row in tile
  int aseg = (tid & 3) * 16;           // 16-half segment
  int brow = tid >> 1;                 // 0..255, weight row (o)
  int bseg = (tid & 1) * 32;           // 32-half segment

  const float*    abase = x + (size_t)tok_s[arow] * D_IN + aseg;
  const _Float16* bbase = wh + (size_t)e * D_OUT * D_IN + (size_t)brow * D_IN + bseg;

  float4 areg[4];

  auto load_stage_a = [&](int kt) {
    const float4* asrc = (const float4*)(abase + kt * K_TILE);
    #pragma unroll
    for (int j = 0; j < 4; ++j) areg[j] = asrc[j];
  };
  auto store_stage_a = [&]() {
    _Float16* ad = &As[arow][aseg];
    #pragma unroll
    for (int j = 0; j < 4; ++j) {
      float4 v = areg[j];
      ad[4*j+0] = (_Float16)v.x; ad[4*j+1] = (_Float16)v.y;
      ad[4*j+2] = (_Float16)v.z; ad[4*j+3] = (_Float16)v.w;
    }
  };
  // async B staging: 4 x 16B per thread, HBM/L2 -> LDS, no VGPR data path.
  // VDST = LDS byte address (low 32 bits of flat shared pointer),
  // VADDR = 64-bit global address, saddr = off. Tracked by ASYNCcnt.
  auto issue_async_b = [&](int kt, int buf) {
    const _Float16* src = bbase + kt * K_TILE;
    _Float16* dst = &Bs[buf][brow][bseg];
    #pragma unroll
    for (int j = 0; j < 4; ++j) {
      unsigned lds_a = (unsigned)(unsigned long long)(dst + j * 8);
      unsigned long long ga = (unsigned long long)(src + j * 8);
      asm volatile("global_load_async_to_lds_b128 %0, %1, off"
                   :: "v"(lds_a), "v"(ga) : "memory");
    }
  };

  int lane = tid & 31;
  int lo = lane & 15, hi = lane >> 4;
  int wv = tid >> 5;
  int ng = wv & 3;                     // N group: 4 sub-tiles
  int mg = wv >> 2;                    // M group: 2 sub-tiles
  int mrow0 = (mg * 2 + 0) * 16 + lo;  // A row for sub-tile 0
  int mrow1 = (mg * 2 + 1) * 16 + lo;  // A row for sub-tile 1

  v8f acc[2][4] = {};
  float bias_r[4];
  #pragma unroll
  for (int c = 0; c < 4; ++c)
    bias_r[c] = bias_proj[e * D_OUT + (ng * 4 + c) * 16 + lo];

  issue_async_b(0, 0);
  load_stage_a(0);
  int buf = 0;
  for (int kt = 0; kt < NKT; ++kt) {
    store_stage_a();
    // this wave's async B transfers for kt have landed:
    asm volatile("s_wait_asynccnt 0x0" ::: "memory");
    __syncthreads();            // publishes everyone's A stores + B async data
    if (kt + 1 < NKT) {
      issue_async_b(kt + 1, buf ^ 1);   // overlaps with WMMAs below
      load_stage_a(kt + 1);             // overlaps with WMMAs below
    }

    #pragma unroll
    for (int kk = 0; kk < K_TILE; kk += 32) {
      // A fragments (ISA 16-bit A layout): lane lo = row M,
      // halves 0..7 = K[kk+hi*8 ..], halves 8..15 = K[kk+16+hi*8 ..]
      V16 a0, a1, b0, b1, b2, b3;
      a0.h[0] = *(const v8h*)&As[mrow0][kk + hi * 8];
      a0.h[1] = *(const v8h*)&As[mrow0][kk + 16 + hi * 8];
      a1.h[0] = *(const v8h*)&As[mrow1][kk + hi * 8];
      a1.h[1] = *(const v8h*)&As[mrow1][kk + 16 + hi * 8];
      // B fragments: lane lo = col N; same K order as A:
      // halves 0..7 = K[kk+hi*8 ..], halves 8..15 = K[kk+16+hi*8 ..] of Bs[buf][o][k]
      {
        const v8h* p0 = (const v8h*)&Bs[buf][(ng * 4 + 0) * 16 + lo][kk + hi * 8];
        const v8h* p1 = (const v8h*)&Bs[buf][(ng * 4 + 1) * 16 + lo][kk + hi * 8];
        const v8h* p2 = (const v8h*)&Bs[buf][(ng * 4 + 2) * 16 + lo][kk + hi * 8];
        const v8h* p3 = (const v8h*)&Bs[buf][(ng * 4 + 3) * 16 + lo][kk + hi * 8];
        b0.h[0] = p0[0]; b0.h[1] = p0[2];
        b1.h[0] = p1[0]; b1.h[1] = p1[2];
        b2.h[0] = p2[0]; b2.h[1] = p2[2];
        b3.h[0] = p3[0]; b3.h[1] = p3[2];
      }
      // 8 WMMAs on 6 preloaded fragments
      acc[0][0] = WMMA_F16(a0.v, b0.v, acc[0][0]);
      acc[0][1] = WMMA_F16(a0.v, b1.v, acc[0][1]);
      acc[0][2] = WMMA_F16(a0.v, b2.v, acc[0][2]);
      acc[0][3] = WMMA_F16(a0.v, b3.v, acc[0][3]);
      acc[1][0] = WMMA_F16(a1.v, b0.v, acc[1][0]);
      acc[1][1] = WMMA_F16(a1.v, b1.v, acc[1][1]);
      acc[1][2] = WMMA_F16(a1.v, b2.v, acc[1][2]);
      acc[1][3] = WMMA_F16(a1.v, b3.v, acc[1][3]);
    }
    __syncthreads();            // As reused next kt; Bs ping-pongs
    buf ^= 1;
  }

  // ---- epilogue: C/D layout VGPR r -> M = r + 8*hi, N = lo ----
  #pragma unroll
  for (int m = 0; m < 2; ++m) {
    #pragma unroll
    for (int c = 0; c < 4; ++c) {
      int o = (ng * 4 + c) * 16 + lo;
      #pragma unroll
      for (int r = 0; r < 8; ++r) {
        int row = (mg * 2 + m) * 16 + hi * 8 + r;
        if (row_base + row < cnt) {
          float v = gate_s[row] * (acc[m][c][r] - bias_r[c]);
          atomicAdd(&out[(size_t)tok_s[row] * D_OUT + o], v);
        }
      }
    }
  }
}

// ---------------------------------------------------------------------------
extern "C" void kernel_launch(void* const* d_in, const int* in_sizes, int n_in,
                              void* d_out, int out_size, void* d_ws, size_t ws_size,
                              hipStream_t stream) {
  const float* x      = (const float*)d_in[0];
  const float* w_gate = (const float*)d_in[1];
  const float* ew     = (const float*)d_in[2];
  const float* eb     = (const float*)d_in[3];
  float* out = (float*)d_out;
  int n_tok = in_sizes[0] / D_IN;

  // workspace layout (all 256B-aligned)
  char* ws = (char*)d_ws;
  size_t off = 0;
  int* counts = (int*)(ws + off);            off += 256;
  int* tok_idx = (int*)(ws + off);           off += (size_t)NEXP * n_tok * sizeof(int);
  float* gate_arr = (float*)(ws + off);      off += (size_t)NEXP * n_tok * sizeof(float);
  _Float16* wh = (_Float16*)(ws + off);      off += (size_t)NEXP * D_OUT * D_IN * sizeof(_Float16);
  float* bias_proj = (float*)(ws + off);     off += (size_t)NEXP * D_OUT * sizeof(float);

  hipMemsetAsync(counts, 0, 256, stream);
  hipMemsetAsync(d_out, 0, (size_t)out_size * sizeof(float), stream); // also zeroes load_loss

  moe_prep<<<NEXP * D_OUT, 256, 0, stream>>>(ew, eb, wh, bias_proj);
  moe_gate<<<n_tok / 256, 256, 0, stream>>>(x, w_gate, counts, tok_idx, gate_arr, n_tok);

  dim3 g((n_tok + M_TILE - 1) / M_TILE, NEXP);
  moe_gemm<<<g, 512, 0, stream>>>(x, wh, bias_proj, counts, tok_idx, gate_arr, out, n_tok);
}
